// GATmodel_57526791963049
// MI455X (gfx1250) — matrix operand order repaired and checked by reading, not compiled, not run on hardware
//
#include <hip/hip_runtime.h>

// ---------------------------------------------------------------------------
// GATv2 (fwd+bwd) x2 layers for MI455X (gfx1250, wave32, WMMA).
// GEMMs run on v_wmma_f32_16x16x32_bf16; edge softmax/aggregation use
// L2-resident float atomics; edge embedding computed on the fly from LDS.
// ---------------------------------------------------------------------------

#define NN   10000      // nodes
#define EE   100000     // edges
#define DIN  128
#define HH   4
#define CC   128
#define HC   512        // H*C
#define ED   16
#define LL   2
#define NEG_SLOPE 0.2f

#define CDIV(a, b) (((a) + (b) - 1) / (b))

typedef __attribute__((ext_vector_type(16))) __bf16 v16bf;
typedef __attribute__((ext_vector_type(8)))  __bf16 v8bf;
typedef __attribute__((ext_vector_type(8)))  float  v8f;

union FragAB { v16bf v; v8bf h[2]; };

__device__ __forceinline__ unsigned short f32_to_bf16(float f) {
  union { float f; unsigned int u; } v; v.f = f;
  unsigned int u = v.u;
  u += 0x7FFFu + ((u >> 16) & 1u);   // round to nearest even
  return (unsigned short)(u >> 16);
}

// ---------------- elementwise helpers ----------------
__global__ void fill_kernel(float* __restrict__ p, float v, int n) {
  int i = blockIdx.x * blockDim.x + threadIdx.x;
  if (i < n) p[i] = v;
}

__global__ void f32_to_bf16_kernel(const float* __restrict__ in,
                                   unsigned short* __restrict__ out, int n) {
  int i = blockIdx.x * blockDim.x + threadIdx.x;
  if (i < n) out[i] = f32_to_bf16(in[i]);
}

// W [K x Nn] f32 row-major  ->  Wt [Nn x K] bf16 row-major (i.e. B transposed)
__global__ void transpose_bf16_kernel(const float* __restrict__ W,
                                      unsigned short* __restrict__ Wt,
                                      int K, int Nn) {
  int i = blockIdx.x * blockDim.x + threadIdx.x;
  if (i >= K * Nn) return;
  int n = i / K;
  int k = i - n * K;
  Wt[i] = f32_to_bf16(W[k * Nn + n]);
}

// concat(fout, bout) -> bf16 [NN x 256]
__global__ void pack_mcat_kernel(const float* __restrict__ f,
                                 const float* __restrict__ b,
                                 unsigned short* __restrict__ out) {
  int i = blockIdx.x * blockDim.x + threadIdx.x;
  if (i >= NN * 256) return;
  int node = i >> 8;
  int c = i & 255;
  float v = (c < CC) ? f[node * CC + c] : b[node * CC + (c - CC)];
  out[i] = f32_to_bf16(v);
}

// mean over heads + bias : agg [NN x 512] -> out [NN x 128]
__global__ void headmean_kernel(const float* __restrict__ agg,
                                const float* __restrict__ bias,
                                float* __restrict__ out) {
  int i = blockIdx.x * blockDim.x + threadIdx.x;
  if (i >= NN * CC) return;
  int n = i >> 7;
  int c = i & 127;
  const float* a = agg + (size_t)n * HC;
  out[i] = 0.25f * (a[c] + a[CC + c] + a[2 * CC + c] + a[3 * CC + c]) + bias[c];
}

// ---------------- WMMA GEMM: C = A(MxK,bf16) * B(KxN) + bias ----------------
// Bt is B transposed: [Nn x K] bf16 row-major. One wave per 16x16 tile.
// A fragment layout (16-bit A 16x32, ISA 7.12.2): lane<16 holds row M=lane,
// K = {kk..kk+7, kk+16..kk+23}; lane>=16 holds K = {kk+8..kk+15, kk+24..kk+31}.
// B fragment: lane<16 holds col N=lane, same K order as A (K = {kk..kk+7, kk+16..kk+23}; lane>=16 K = {kk+8..kk+15, kk+24..kk+31}).
template <bool RELU>
__global__ __launch_bounds__(256) void
gemm_bf16_kernel(const unsigned short* __restrict__ A,
                 const unsigned short* __restrict__ Bt,
                 const float* __restrict__ bias,
                 float* __restrict__ Cmat,
                 int M, int Nn, int K) {
  const int lane = threadIdx.x & 31;
  const int wavesPerBlock = blockDim.x >> 5;
  const int tile = blockIdx.x * wavesPerBlock + (threadIdx.x >> 5);
  const int ntn = Nn >> 4;
  const int total = (M >> 4) * ntn;
  if (tile >= total) return;                 // wave-uniform
  const int tm = tile / ntn;
  const int tn = tile - tm * ntn;

  const int halfsel = lane >> 4;             // 0 | 1
  const int l15 = lane & 15;
  const size_t arow = (size_t)(tm * 16 + l15) * K;
  const size_t brow = (size_t)(tn * 16 + l15) * K;

  v8f acc = {};
  for (int kk = 0; kk < K; kk += 32) {
    FragAB fa, fb;
    const int ka = kk + (halfsel ? 8 : 0);
    fa.h[0] = *(const v8bf*)(A + arow + ka);
    fa.h[1] = *(const v8bf*)(A + arow + ka + 16);
    const int kb = kk + (halfsel ? 8 : 0);
    fb.h[0] = *(const v8bf*)(Bt + brow + kb);
    fb.h[1] = *(const v8bf*)(Bt + brow + kb + 16);
    acc = __builtin_amdgcn_wmma_f32_16x16x32_bf16(
        false, fa.v, false, fb.v, (short)0, acc, false, false);
  }

  const int cn = tn * 16 + l15;
  const int rbase = tm * 16 + (halfsel ? 8 : 0);
  const float bv = bias[cn];
#pragma unroll
  for (int r = 0; r < 8; ++r) {
    float v = acc[r] + bv;
    if (RELU) v = fmaxf(v, 0.f);
    Cmat[(size_t)(rbase + r) * Nn + cn] = v;
  }
}

// ---------------- edge pass 1: logits + running max --------------------------
// One wave per edge; lane covers 16 channels hc = lane*16..+15 (one head per
// 8-lane group). ee computed on the fly from We held in LDS.
__global__ __launch_bounds__(256) void
edge_logits_kernel(const int* __restrict__ src, const int* __restrict__ dst,
                   const float* __restrict__ eattr,  // [EE x 16]
                   const float* __restrict__ We,     // [16 x 512]
                   const float* __restrict__ att,    // [512] (H*C flat)
                   const float* __restrict__ xl, const float* __restrict__ xr,
                   float* __restrict__ logits,       // [EE x 4]
                   float* __restrict__ amax) {       // [NN x 4]
  __shared__ float sWe[ED * HC];
  __shared__ float sAtt[HC];
  for (int i = threadIdx.x; i < ED * HC; i += blockDim.x) sWe[i] = We[i];
  for (int i = threadIdx.x; i < HC; i += blockDim.x) sAtt[i] = att[i];
  __syncthreads();

  const int lane = threadIdx.x & 31;
  const int wave = blockIdx.x * (blockDim.x >> 5) + (threadIdx.x >> 5);
  const int nwaves = gridDim.x * (blockDim.x >> 5);
  const int hc0 = lane * 16;
  const int h = lane >> 3;

  for (int e = wave; e < EE; e += nwaves) {
    const int s = src[e], d = dst[e];
    float ea[ED];
#pragma unroll
    for (int k = 0; k < ED; ++k) ea[k] = eattr[e * ED + k];

    const float4* pxl = (const float4*)(xl + (size_t)s * HC + hc0);
    const float4* pxr = (const float4*)(xr + (size_t)d * HC + hc0);
    float part = 0.f;
#pragma unroll
    for (int j4 = 0; j4 < 4; ++j4) {
      float4 vl = pxl[j4];
      float4 vr = pxr[j4];
      float vls[4] = {vl.x, vl.y, vl.z, vl.w};
      float vrs[4] = {vr.x, vr.y, vr.z, vr.w};
#pragma unroll
      for (int t = 0; t < 4; ++t) {
        const int hc = hc0 + j4 * 4 + t;
        float ee = 0.f;
#pragma unroll
        for (int k = 0; k < ED; ++k) ee = fmaf(ea[k], sWe[k * HC + hc], ee);
        float z = vls[t] + vrs[t] + ee;
        z = (z > 0.f) ? z : NEG_SLOPE * z;
        part = fmaf(z, sAtt[hc], part);
      }
    }
    part += __shfl_xor(part, 1);
    part += __shfl_xor(part, 2);
    part += __shfl_xor(part, 4);
    if ((lane & 7) == 0) {
      logits[e * HH + h] = part;
      atomicMax(&amax[d * HH + h], part);
    }
  }
}

// ---------------- edge pass 2: exp + denom -----------------------------------
__global__ void edge_exp_kernel(const int* __restrict__ dst,
                                float* __restrict__ logits,
                                const float* __restrict__ amax,
                                float* __restrict__ denom) {
  int i = blockIdx.x * blockDim.x + threadIdx.x;
  if (i >= EE * HH) return;
  int e = i >> 2, h = i & 3;
  int d = dst[e];
  float ex = __expf(logits[i] - amax[d * HH + h]);
  logits[i] = ex;                 // reuse buffer for exp values
  atomicAdd(&denom[d * HH + h], ex);
}

// ---------------- edge pass 3: weighted scatter ------------------------------
__global__ __launch_bounds__(256) void
edge_agg_kernel(const int* __restrict__ src, const int* __restrict__ dst,
                const float* __restrict__ exv,    // [EE x 4]
                const float* __restrict__ denom,  // [NN x 4]
                const float* __restrict__ xl,     // [NN x 512]
                float* __restrict__ agg) {        // [NN x 512]
  const int lane = threadIdx.x & 31;
  const int wave = blockIdx.x * (blockDim.x >> 5) + (threadIdx.x >> 5);
  const int nwaves = gridDim.x * (blockDim.x >> 5);
  const int hc0 = lane * 16;
  const int h = lane >> 3;

  for (int e = wave; e < EE; e += nwaves) {
    const int s = src[e], d = dst[e];
    const float alpha = exv[e * HH + h] / denom[d * HH + h];
    const float4* pxl = (const float4*)(xl + (size_t)s * HC + hc0);
    float* ag = agg + (size_t)d * HC + hc0;
#pragma unroll
    for (int j4 = 0; j4 < 4; ++j4) {
      float4 v = pxl[j4];
      atomicAdd(ag + j4 * 4 + 0, v.x * alpha);
      atomicAdd(ag + j4 * 4 + 1, v.y * alpha);
      atomicAdd(ag + j4 * 4 + 2, v.z * alpha);
      atomicAdd(ag + j4 * 4 + 3, v.w * alpha);
    }
  }
}

// ---------------- batch norm ----------------
__global__ __launch_bounds__(256) void
bn_stats_kernel(const float* __restrict__ m, float* __restrict__ mu,
                float* __restrict__ rstd) {
  __shared__ float s1[256];
  __shared__ float s2[256];
  const int c = blockIdx.x;     // 128 blocks
  float a = 0.f, b = 0.f;
  for (int n = threadIdx.x; n < NN; n += blockDim.x) {
    float v = m[(size_t)n * CC + c];
    a += v;
    b += v * v;
  }
  s1[threadIdx.x] = a;
  s2[threadIdx.x] = b;
  __syncthreads();
  for (int off = 128; off > 0; off >>= 1) {
    if ((int)threadIdx.x < off) {
      s1[threadIdx.x] += s1[threadIdx.x + off];
      s2[threadIdx.x] += s2[threadIdx.x + off];
    }
    __syncthreads();
  }
  if (threadIdx.x == 0) {
    float mean = s1[0] / (float)NN;
    float var = s2[0] / (float)NN - mean * mean;
    mu[c] = mean;
    rstd[c] = rsqrtf(var + 1e-5f);
  }
}

__global__ void bn_apply_kernel(const float* __restrict__ m,
                                const float* __restrict__ gamma,
                                const float* __restrict__ beta,
                                const float* __restrict__ mu,
                                const float* __restrict__ rstd,
                                float* __restrict__ out) {
  int i = blockIdx.x * blockDim.x + threadIdx.x;
  if (i >= NN * CC) return;
  int c = i & 127;
  float y = gamma[c] * (m[i] - mu[c]) * rstd[c] + beta[c];
  out[i] = fmaxf(y, 0.f);
}

// ---------------------------------------------------------------------------
extern "C" void kernel_launch(void* const* d_in, const int* in_sizes, int n_in,
                              void* d_out, int out_size, void* d_ws, size_t ws_size,
                              hipStream_t stream) {
  (void)in_sizes; (void)n_in; (void)out_size; (void)ws_size;

  const float* x      = (const float*)d_in[0];
  const int*   fwd    = (const int*)d_in[1];
  const int*   bwd    = (const int*)d_in[2];
  const float* eattr  = (const float*)d_in[3];
  // per-direction param tables: [0]=fwd(idx 4..10), [1]=bwd(idx 11..17)
  const float* Wl[2]   = {(const float*)d_in[4],  (const float*)d_in[11]};
  const float* bl[2]   = {(const float*)d_in[5],  (const float*)d_in[12]};
  const float* Wr[2]   = {(const float*)d_in[6],  (const float*)d_in[13]};
  const float* br[2]   = {(const float*)d_in[7],  (const float*)d_in[14]};
  const float* We[2]   = {(const float*)d_in[8],  (const float*)d_in[15]};
  const float* att[2]  = {(const float*)d_in[9],  (const float*)d_in[16]};
  const float* bias[2] = {(const float*)d_in[10], (const float*)d_in[17]};
  const float* Wm1   = (const float*)d_in[18];
  const float* bm1   = (const float*)d_in[19];
  const float* Wm2   = (const float*)d_in[20];
  const float* bm2   = (const float*)d_in[21];
  const float* gamma = (const float*)d_in[22];
  const float* beta  = (const float*)d_in[23];

  // ---- workspace layout (bytes) ----
  char* ws = (char*)d_ws;
  float*          hbuf   = (float*)(ws + 0);                    //  5,120,000  (m2 / h)
  unsigned short* hbf    = (unsigned short*)(ws + 5120000);     //  2,560,000
  unsigned short* m1bf   = (unsigned short*)(ws + 7680000);     //  2,560,000
  unsigned short* mcatbf = (unsigned short*)(ws + 10240000);    //  5,120,000
  float*          xl     = (float*)(ws + 15360000);             // 20,480,000
  float*          xr     = (float*)(ws + 35840000);             // 20,480,000
  float*          agg    = (float*)(ws + 56320000);             // 20,480,000
  float*          fout   = (float*)(ws + 76800000);             //  5,120,000
  float*          bout   = (float*)(ws + 81920000);             //  5,120,000
  float*          m1     = (float*)(ws + 87040000);             //  5,120,000
  float*          logits = (float*)(ws + 92160000);             //  1,600,000
  float*          amax   = (float*)(ws + 93760000);             //    160,000
  float*          denom  = (float*)(ws + 93920000);             //    160,000
  float*          muB    = (float*)(ws + 94080000);             //        512
  float*          rstdB  = (float*)(ws + 94080512);             //        512
  unsigned short* WtA    = (unsigned short*)(ws + 94081024);    //    131,072
  unsigned short* WtB    = (unsigned short*)(ws + 94212096);    //    131,072
  unsigned short* WtM1   = (unsigned short*)(ws + 94343168);    //     65,536
  unsigned short* WtM2   = (unsigned short*)(ws + 94408704);    //     32,768

  const float NEG_INF = -__builtin_inff();

  for (int l = 0; l < LL; ++l) {
    const float* hsrc = (l == 0) ? x : hbuf;
    f32_to_bf16_kernel<<<CDIV(NN * DIN, 256), 256, 0, stream>>>(hsrc, hbf, NN * DIN);

    for (int dir = 0; dir < 2; ++dir) {
      const int* ei  = (dir == 0) ? fwd : bwd;
      const int* src = ei;
      const int* dst = ei + EE;
      const float* Wl_l   = Wl[dir]   + (size_t)l * DIN * HC;
      const float* bl_l   = bl[dir]   + (size_t)l * HC;
      const float* Wr_l   = Wr[dir]   + (size_t)l * DIN * HC;
      const float* br_l   = br[dir]   + (size_t)l * HC;
      const float* We_l   = We[dir]   + (size_t)l * ED * HC;
      const float* att_l  = att[dir]  + (size_t)l * HC;
      const float* bias_l = bias[dir] + (size_t)l * CC;

      // xl = h @ Wl + bl ; xr = h @ Wr + br   (WMMA bf16)
      transpose_bf16_kernel<<<CDIV(DIN * HC, 256), 256, 0, stream>>>(Wl_l, WtA, DIN, HC);
      transpose_bf16_kernel<<<CDIV(DIN * HC, 256), 256, 0, stream>>>(Wr_l, WtB, DIN, HC);
      gemm_bf16_kernel<false><<<(NN / 16) * (HC / 16) / 8, 256, 0, stream>>>(
          hbf, WtA, bl_l, xl, NN, HC, DIN);
      gemm_bf16_kernel<false><<<(NN / 16) * (HC / 16) / 8, 256, 0, stream>>>(
          hbf, WtB, br_l, xr, NN, HC, DIN);

      // scatter softmax + aggregation
      fill_kernel<<<CDIV(NN * HH, 256), 256, 0, stream>>>(amax, NEG_INF, NN * HH);
      fill_kernel<<<CDIV(NN * HH, 256), 256, 0, stream>>>(denom, 0.f, NN * HH);
      fill_kernel<<<CDIV(NN * HC, 256), 256, 0, stream>>>(agg, 0.f, NN * HC);
      edge_logits_kernel<<<2048, 256, 0, stream>>>(src, dst, eattr, We_l, att_l,
                                                   xl, xr, logits, amax);
      edge_exp_kernel<<<CDIV(EE * HH, 256), 256, 0, stream>>>(dst, logits, amax, denom);
      edge_agg_kernel<<<2048, 256, 0, stream>>>(src, dst, logits, denom, xl, agg);
      headmean_kernel<<<CDIV(NN * CC, 256), 256, 0, stream>>>(
          agg, bias_l, (dir == 0) ? fout : bout);
    }

    // merge MLP: relu(concat @ Wm1 + bm1) @ Wm2 + bm2
    pack_mcat_kernel<<<CDIV(NN * 256, 256), 256, 0, stream>>>(fout, bout, mcatbf);
    transpose_bf16_kernel<<<CDIV(256 * CC, 256), 256, 0, stream>>>(
        Wm1 + (size_t)l * 256 * CC, WtM1, 256, CC);
    gemm_bf16_kernel<true><<<(NN / 16) * (CC / 16) / 8, 256, 0, stream>>>(
        mcatbf, WtM1, bm1 + (size_t)l * CC, m1, NN, CC, 256);
    f32_to_bf16_kernel<<<CDIV(NN * CC, 256), 256, 0, stream>>>(m1, m1bf, NN * CC);
    transpose_bf16_kernel<<<CDIV(CC * CC, 256), 256, 0, stream>>>(
        Wm2 + (size_t)l * CC * CC, WtM2, CC, CC);
    gemm_bf16_kernel<false><<<(NN / 16) * (CC / 16) / 8, 256, 0, stream>>>(
        m1bf, WtM2, bm2 + (size_t)l * CC, hbuf, NN, CC, CC);

    // batch norm (training stats) + relu
    bn_stats_kernel<<<CC, 256, 0, stream>>>(hbuf, muB, rstdB);
    bn_apply_kernel<<<CDIV(NN * CC, 256), 256, 0, stream>>>(
        hbuf, gamma + (size_t)l * CC, beta + (size_t)l * CC, muB, rstdB,
        (l == LL - 1) ? (float*)d_out : hbuf);
  }
}
